// GCNModel_52982716564271
// MI455X (gfx1250) — matrix operand order repaired and checked by reading, not compiled, not run on hardware
//
#include <hip/hip_runtime.h>

// ---------------------------------------------------------------------------
// GCN forward on gfx1250 (MI455X).  GEMM via v_wmma_f32_16x16x32_bf16 with
// hi/lo bf16 split of fp32 operands (3 WMMAs per k-step) for fp32-class
// accuracy.  A operands loaded as packed uint4 (ISA 16-bit A lane layout is
// K-contiguous per VGPR quad); B operands loaded like A (two uint4 runs) from
// pre-transposed weights.  WMMAs issued in interleaved passes over the 4
// accumulators so adjacent WMMAs are independent (no D->C serialization, no
// hazard NOPs).  Scatter via hw f32 global atomics.  wave32 throughout.
// ---------------------------------------------------------------------------

typedef __attribute__((ext_vector_type(16))) __bf16 v16bf;
typedef __attribute__((ext_vector_type(8)))  float  v8f;

static __device__ __forceinline__ unsigned short f32_to_bf16_rne(float f) {
    unsigned int u = __builtin_bit_cast(unsigned int, f);
    u += 0x7fffu + ((u >> 16) & 1u);          // round to nearest even
    return (unsigned short)(u >> 16);
}
static __device__ __forceinline__ float bf16_to_f32(unsigned short h) {
    unsigned int u = ((unsigned int)h) << 16;
    return __builtin_bit_cast(float, u);
}
static __device__ __forceinline__ void atomic_add_f32(float* p, float v) {
    __hip_atomic_fetch_add(p, v, __ATOMIC_RELAXED, __HIP_MEMORY_SCOPE_AGENT);
}

// A-operand: lane's 16 bf16 = two contiguous 8-element (16B) runs, 16 K apart
static __device__ __forceinline__ v16bf load_a16(const unsigned short* p) {
    union { uint4 q[2]; v16bf v; } u;
    u.q[0] = *(const uint4*)(p);        // K = kb .. kb+7       -> elems 0..7
    u.q[1] = *(const uint4*)(p + 16);   // K = 16+kb .. 16+kb+7 -> elems 8..15
    return u.v;
}

// ---------------------------------------------------------------------------
// small utility kernels
// ---------------------------------------------------------------------------
__global__ __launch_bounds__(256) void k_zero_f32(float* p, int n) {
    int t = blockIdx.x * blockDim.x + threadIdx.x;
    if (t < n) p[t] = 0.0f;
}

__global__ __launch_bounds__(256) void k_deg_accum(const int* __restrict__ dst,
                                                   float* __restrict__ deg, int E) {
    int t = blockIdx.x * blockDim.x + threadIdx.x;
    if (t < E) atomic_add_f32(&deg[dst[t]], 1.0f);
}

__global__ __launch_bounds__(256) void k_deg_to_dinv(float* __restrict__ d, int n) {
    int t = blockIdx.x * blockDim.x + threadIdx.x;
    if (t < n) d[t] = __frsqrt_rn(d[t] + 1.0f);   // +1 self loop; deg >= 1
}

// split fp32 -> (hi, lo) bf16 pair, row-major (activations):  x ~= hi + lo
__global__ __launch_bounds__(256) void k_split_bf16(const float* __restrict__ x,
                                                    unsigned short* __restrict__ hi,
                                                    unsigned short* __restrict__ lo, int n) {
    int t = blockIdx.x * blockDim.x + threadIdx.x;
    if (t >= n) return;
    float v = x[t];
    unsigned short h = f32_to_bf16_rne(v);
    hi[t] = h;
    lo[t] = f32_to_bf16_rne(v - bf16_to_f32(h));
}

// split + transpose weights: W[fin][fo] -> WT{hi,lo}[fo][fin] (K contiguous)
__global__ __launch_bounds__(256) void k_wsplit_t(const float* __restrict__ W,
                                                  unsigned short* __restrict__ WThi,
                                                  unsigned short* __restrict__ WTlo,
                                                  int fin, int fo) {
    int t = blockIdx.x * blockDim.x + threadIdx.x;
    if (t >= fin * fo) return;
    int k = t / fo, n = t - k * fo;
    float v = W[t];
    unsigned short h = f32_to_bf16_rne(v);
    WThi[(size_t)n * fin + k] = h;
    WTlo[(size_t)n * fin + k] = f32_to_bf16_rne(v - bf16_to_f32(h));
}

// ---------------------------------------------------------------------------
// GEMM: C[M,NF] = A[M,K] * Wt^T, fp32 accumulated from bf16 hi/lo split.
// One wave32 computes a 16 x (16*NT) tile; A loads reused across NT n-tiles.
// A lane layout (ISA 7.12.2, 16-bit A 16x32): row = tm+(L&15), kb=(L>=16)?8:0,
//   elems 0..7 -> K=kb..kb+7, elems 8..15 -> K=16+kb..16+kb+7 (contiguous).
// B lane layout (32x16): col = tn+(L&15), kb=(L>=16)?8:0,
//   elems 0..7 -> K=kb..kb+7, elems 8..15 -> K=16+kb..16+kb+7 (same as A).
// C/D: vgpr r -> row tm + r + 8*(L>=16), col tn + (L&15).
// ---------------------------------------------------------------------------
template <int K, int NF, int NT>
__global__ __launch_bounds__(32) void k_gemm_wmma(
    const unsigned short* __restrict__ Ahi, const unsigned short* __restrict__ Alo,
    const unsigned short* __restrict__ WThi, const unsigned short* __restrict__ WTlo,
    float* __restrict__ C)
{
    const int tm   = blockIdx.x * 16;
    const int tn0  = blockIdx.y * (16 * NT);
    const int lane = threadIdx.x;
    const int half = lane >> 4;        // 0 | 1
    const int l15  = lane & 15;
    const int akb  = half * 8;
    const int bkb  = half * 8;

    const unsigned short* arh = Ahi + (size_t)(tm + l15) * K;
    const unsigned short* arl = Alo + (size_t)(tm + l15) * K;

    v8f acc[NT];
#pragma unroll
    for (int nt = 0; nt < NT; ++nt) acc[nt] = (v8f){};

#pragma unroll
    for (int kc = 0; kc < K; kc += 32) {
        const v16bf ah = load_a16(arh + kc + akb);
        const v16bf al = load_a16(arl + kc + akb);

        v16bf bh[NT], bl[NT];
#pragma unroll
        for (int nt = 0; nt < NT; ++nt) {
            const size_t bro = (size_t)(tn0 + nt * 16 + l15) * K + kc + bkb;
            bh[nt] = load_a16(WThi + bro);
            bl[nt] = load_a16(WTlo + bro);
        }
        // Three interleaved passes: adjacent WMMAs hit different accumulators,
        // so no D->C dependency between consecutive matrix ops.
#pragma unroll
        for (int nt = 0; nt < NT; ++nt)
            acc[nt] = __builtin_amdgcn_wmma_f32_16x16x32_bf16(false, ah, false, bh[nt], (short)0, acc[nt], false, false);
#pragma unroll
        for (int nt = 0; nt < NT; ++nt)
            acc[nt] = __builtin_amdgcn_wmma_f32_16x16x32_bf16(false, ah, false, bl[nt], (short)0, acc[nt], false, false);
#pragma unroll
        for (int nt = 0; nt < NT; ++nt)
            acc[nt] = __builtin_amdgcn_wmma_f32_16x16x32_bf16(false, al, false, bh[nt], (short)0, acc[nt], false, false);
    }

    float* crow = C + (size_t)(tm + 8 * half) * NF + l15;
#pragma unroll
    for (int nt = 0; nt < NT; ++nt)
#pragma unroll
        for (int r = 0; r < 8; ++r)
            crow[(size_t)r * NF + tn0 + nt * 16] = acc[nt][r];
}

// Agg[i][f] = b[f] + G[i][f] * dinv[i]^2   (self-loop term + bias, full init)
__global__ __launch_bounds__(256) void k_agg_init(const float* __restrict__ G,
                                                  const float* __restrict__ dinv,
                                                  const float* __restrict__ bias,
                                                  float* __restrict__ Agg, int n, int F) {
    int t = blockIdx.x * blockDim.x + threadIdx.x;
    if (t >= n) return;
    int i = t / F, f = t - i * F;
    float di = dinv[i];
    Agg[t] = bias[f] + G[t] * di * di;
}

// Agg[dst] += G[src] * dinv[src]*dinv[dst]; F/4 lanes per edge, float4 gather
__global__ __launch_bounds__(256) void k_edge_scatter(const int* __restrict__ src,
                                                      const int* __restrict__ dst,
                                                      const float* __restrict__ dinv,
                                                      const float* __restrict__ G,
                                                      float* __restrict__ Agg,
                                                      int E, int F) {
    int t   = blockIdx.x * blockDim.x + threadIdx.x;
    int per = F >> 2;                        // threads per edge
    int e   = t / per;
    if (e >= E) return;
    int c   = (t - e * per) * 4;
    int s = src[e], d = dst[e];
    float w = dinv[s] * dinv[d];
    const float4 g = *(const float4*)(G + (size_t)s * F + c);
    float* out = Agg + (size_t)d * F + c;
    atomic_add_f32(out + 0, g.x * w);
    atomic_add_f32(out + 1, g.y * w);
    atomic_add_f32(out + 2, g.z * w);
    atomic_add_f32(out + 3, g.w * w);
}

__global__ __launch_bounds__(256) void k_relu(float* __restrict__ p, int n) {
    int t = blockIdx.x * blockDim.x + threadIdx.x;
    if (t < n) p[t] = fmaxf(p[t], 0.0f);
}

__global__ __launch_bounds__(256) void k_pool_accum(const float* __restrict__ Agg,
                                                    const int* __restrict__ batch,
                                                    float* __restrict__ sums,
                                                    float* __restrict__ cnts,
                                                    int Nn, int F) {
    int t = blockIdx.x * blockDim.x + threadIdx.x;
    if (t >= Nn * F) return;
    int i = t / F, f = t - i * F;
    int g = batch[i];
    atomic_add_f32(&sums[g * F + f], Agg[(size_t)i * F + f]);
    if (f == 0) atomic_add_f32(&cnts[g], 1.0f);
}

__global__ __launch_bounds__(256) void k_pool_div(const float* __restrict__ sums,
                                                  const float* __restrict__ cnts,
                                                  float* __restrict__ out, int n, int F) {
    int t = blockIdx.x * blockDim.x + threadIdx.x;
    if (t >= n) return;
    out[t] = sums[t] / fmaxf(cnts[t / F], 1.0f);
}

// ---------------------------------------------------------------------------
// host-side orchestration
// ---------------------------------------------------------------------------
static inline size_t align256(size_t x) { return (x + 255) & ~(size_t)255; }

extern "C" void kernel_launch(void* const* d_in, const int* in_sizes, int n_in,
                              void* d_out, int out_size, void* d_ws, size_t ws_size,
                              hipStream_t stream) {
    const int FIN = 128, NG = 64;
    const float* x     = (const float*)d_in[0];
    const int*   ei    = (const int*)d_in[1];
    const int*   batch = (const int*)d_in[2];
    const float* W[3]  = {(const float*)d_in[3], (const float*)d_in[5], (const float*)d_in[7]};
    const float* B[3]  = {(const float*)d_in[4], (const float*)d_in[6], (const float*)d_in[8]};
    const int Fout[3]  = {128, 128, 64};

    const int N = in_sizes[0] / FIN;     // 100000 (multiple of 16)
    const int E = in_sizes[1] / 2;       // 1.6M
    const int* src = ei;
    const int* dst = ei + E;

    // workspace carve-up
    char* ws = (char*)d_ws;
    size_t off = 0;
    float* dinv = (float*)(ws + off);                  off += align256((size_t)N * 4);
    float* G    = (float*)(ws + off);                  off += align256((size_t)N * 128 * 4);
    float* Agg  = (float*)(ws + off);                  off += align256((size_t)N * 128 * 4);
    unsigned short* Hhi = (unsigned short*)(ws + off); off += align256((size_t)N * 128 * 2);
    unsigned short* Hlo = (unsigned short*)(ws + off); off += align256((size_t)N * 128 * 2);
    unsigned short* Whi = (unsigned short*)(ws + off); off += align256((size_t)128 * 128 * 2);
    unsigned short* Wlo = (unsigned short*)(ws + off); off += align256((size_t)128 * 128 * 2);
    float* psum = (float*)(ws + off);                  off += align256((size_t)NG * 64 * 4);
    float* pcnt = (float*)(ws + off);                  off += align256((size_t)NG * 4);
    (void)ws_size;

    const int TB = 256;
    auto blocks = [](long long n, int tb) { return (unsigned)((n + tb - 1) / tb); };

    // 1) symmetric normalization: dinv = rsqrt(deg_in + 1)
    k_zero_f32<<<blocks(N, TB), TB, 0, stream>>>(dinv, N);
    k_deg_accum<<<blocks(E, TB), TB, 0, stream>>>(dst, dinv, E);
    k_deg_to_dinv<<<blocks(N, TB), TB, 0, stream>>>(dinv, N);

    // 2) three GCN layers
    const float* layer_in = x;
    int fin = FIN;
    for (int l = 0; l < 3; ++l) {
        const int fo = Fout[l];
        const long long nin  = (long long)N * fin;
        const long long nout = (long long)N * fo;

        // bf16 hi/lo splits: activations row-major, weights transposed
        k_split_bf16<<<blocks(nin, TB), TB, 0, stream>>>(layer_in, Hhi, Hlo, (int)nin);
        k_wsplit_t<<<blocks((long long)fin * fo, TB), TB, 0, stream>>>(W[l], Whi, Wlo, fin, fo);

        // G = H @ W   (WMMA, 16 x 64 tile per wave32)
        if (fo == 128) {
            dim3 g((unsigned)(N / 16), 2);
            k_gemm_wmma<128, 128, 4><<<g, 32, 0, stream>>>(Hhi, Hlo, Whi, Wlo, G);
        } else {
            dim3 g((unsigned)(N / 16), 1);
            k_gemm_wmma<128, 64, 4><<<g, 32, 0, stream>>>(Hhi, Hlo, Whi, Wlo, G);
        }

        // Agg = bias + self-loop, then scatter edges, then ReLU (layers 0,1)
        k_agg_init<<<blocks(nout, TB), TB, 0, stream>>>(G, dinv, B[l], Agg, (int)nout, fo);
        k_edge_scatter<<<blocks((long long)E * (fo >> 2), TB), TB, 0, stream>>>(
            src, dst, dinv, G, Agg, E, fo);
        if (l < 2)
            k_relu<<<blocks(nout, TB), TB, 0, stream>>>(Agg, (int)nout);

        layer_in = Agg;
        fin = fo;
    }

    // 3) global mean pool over graphs
    k_zero_f32<<<blocks(NG * 64, TB), TB, 0, stream>>>(psum, NG * 64);
    k_zero_f32<<<blocks(NG, TB), TB, 0, stream>>>(pcnt, NG);
    k_pool_accum<<<blocks((long long)N * 64, TB), TB, 0, stream>>>(Agg, batch, psum, pcnt, N, 64);
    k_pool_div<<<blocks(NG * 64, TB), TB, 0, stream>>>(psum, pcnt, (float*)d_out, NG * 64, 64);
}
